// RNN_26714696581271
// MI455X (gfx1250) — matrix operand order repaired and checked by reading, not compiled, not run on hardware
//
#include <hip/hip_runtime.h>
#include <hip/hip_bf16.h>

// ---------------------------------------------------------------------------
// 3-layer GRU sampler, persistent kernel, bf16 WMMA (gfx1250 / MI455X).
// Each block = 32 batch rows, 16 wave32s (512 thr); loops over 140 timesteps.
// Weights pre-converted/swizzled into d_ws as per-lane WMMA B fragments.
// Wave(mg,wg): mg = m-tile (rows mg*16..+16), wg = 64-column slice of H.
// ---------------------------------------------------------------------------

typedef __bf16 bf16_t;
typedef __attribute__((ext_vector_type(16))) __bf16 v16bf;
typedef __attribute__((ext_vector_type(8)))  __bf16 v8bf;
typedef __attribute__((ext_vector_type(8)))  float   v8f;

#define BATCH   2048
#define EMBD    128
#define HID     512
#define H3      1536
#define VOC     32
#define MAXLEN  140
#define EOS_TOK 1
#define MROWS   32                 // batch rows per block
#define NTHR    512                // 16 waves
#define HSTR    (HID + 8)          // padded LDS row stride (bf16 elems)
#define XSTR    (EMBD + 8)
#define VSTR    (VOC + 1)

__device__ __forceinline__ float sigmoidf_(float x) {
    return 1.0f / (1.0f + __expf(-x));
}

// A fragment (16x32 bf16) from an LDS tile (base already offset to the
// wave's 16-row m-group), row-major with stride S.
__device__ __forceinline__ v16bf lds_a_frag(const bf16_t* x, int S, int kt, int lane) {
    const int m = lane & 15, half = lane >> 4;
    const bf16_t* p = x + m * S + kt * 32 + half * 16;
    v8bf a0 = *(const v8bf*)p;
    v8bf a1 = *(const v8bf*)(p + 8);
    return __builtin_shufflevector(a0, a1, 0,1,2,3,4,5,6,7,8,9,10,11,12,13,14,15);
}

// B fragment (32x16 bf16) from pre-swizzled weights: fragment for tile
// (nTile,kt) stored as 32 lanes x 16 bf16 contiguous -> coalesced 32B/lane.
__device__ __forceinline__ v16bf g_b_frag(const bf16_t* w, int nTile, int kt,
                                          int nKT, int lane) {
    return *(const v16bf*)(w + (((size_t)nTile * nKT + kt) * 32 + lane) * 16);
}

__device__ __forceinline__ v8f wmma_bf(v16bf a, v16bf b, v8f c) {
    return __builtin_amdgcn_wmma_f32_16x16x32_bf16(false, a, false, b,
                                                   (short)0, c, false, false);
}

// One GRU layer step for one wave: 16 rows (xb/hb pre-offset) x 64 columns.
// r/z gates accumulate both GEMMs into one acc; "new" gate keeps inn/hn split.
template <int KIN, int XS>
__device__ __forceinline__ void gru_layer(const bf16_t* __restrict__ xb,
                                          bf16_t* __restrict__ hb,
                                          const bf16_t* __restrict__ wih,
                                          const bf16_t* __restrict__ whh,
                                          const float* __restrict__ bih,
                                          const float* __restrict__ bhh,
                                          int wg, int lane) {
    constexpr int nKTi = KIN / 32;
    const v8f zero = {0.f,0.f,0.f,0.f,0.f,0.f,0.f,0.f};
    v8f accR[4], accZ[4], accI[4], accH[4];
#pragma unroll
    for (int nt = 0; nt < 4; ++nt) {
        accR[nt] = zero; accZ[nt] = zero; accI[nt] = zero; accH[nt] = zero;
    }
    // Phase 1: gi = x @ wih^T  (r,z,inn)
#pragma unroll 2
    for (int kt = 0; kt < nKTi; ++kt) {
        v16bf a = lds_a_frag(xb, XS, kt, lane);
#pragma unroll
        for (int nt = 0; nt < 4; ++nt) {
            const int nb = wg * 4 + nt;
            accR[nt] = wmma_bf(a, g_b_frag(wih, nb,      kt, nKTi, lane), accR[nt]);
            accZ[nt] = wmma_bf(a, g_b_frag(wih, 32 + nb, kt, nKTi, lane), accZ[nt]);
            accI[nt] = wmma_bf(a, g_b_frag(wih, 64 + nb, kt, nKTi, lane), accI[nt]);
        }
    }
    // Phase 2: gh = h @ whh^T  (r,z fused into accR/accZ; hn separate)
#pragma unroll 2
    for (int kt = 0; kt < 16; ++kt) {
        if (kt + 1 < 16)  // pull next K-slice of whh toward the WGP
            __builtin_prefetch(whh + (((size_t)(wg * 4) * 16) + kt + 1) * 512, 0, 1);
        v16bf a = lds_a_frag(hb, HSTR, kt, lane);
#pragma unroll
        for (int nt = 0; nt < 4; ++nt) {
            const int nb = wg * 4 + nt;
            accR[nt] = wmma_bf(a, g_b_frag(whh, nb,      kt, 16, lane), accR[nt]);
            accZ[nt] = wmma_bf(a, g_b_frag(whh, 32 + nb, kt, 16, lane), accZ[nt]);
            accH[nt] = wmma_bf(a, g_b_frag(whh, 64 + nb, kt, 16, lane), accH[nt]);
        }
    }
    __syncthreads();  // everyone done reading h as A operand

    const int nlo = lane & 15, half = lane >> 4;
#pragma unroll
    for (int nt = 0; nt < 4; ++nt) {
        const int col = wg * 64 + nt * 16 + nlo;
        const float brI = bih[col],         brH = bhh[col];
        const float bzI = bih[HID + col],   bzH = bhh[HID + col];
        const float bnI = bih[2*HID + col], bnH = bhh[2*HID + col];
#pragma unroll
        for (int r8 = 0; r8 < 8; ++r8) {
            const int m = half * 8 + r8;
            const float r = sigmoidf_(accR[nt][r8] + brI + brH);
            const float z = sigmoidf_(accZ[nt][r8] + bzI + bzH);
            const float n = tanhf(accI[nt][r8] + bnI + r * (accH[nt][r8] + bnH));
            const float ho = (float)hb[m * HSTR + col];
            hb[m * HSTR + col] = (bf16_t)((1.0f - z) * n + z * ho);
        }
    }
    __syncthreads();
}

__device__ __forceinline__ unsigned mix32(unsigned x) {
    x ^= x >> 16; x *= 0x7feb352du;
    x ^= x >> 15; x *= 0x846ca68bu;
    x ^= x >> 16;
    return x;
}

__global__ __launch_bounds__(NTHR)
void gru_rnn_persistent(const int* __restrict__ obs,
                        const bf16_t* __restrict__ embb,
                        const bf16_t* __restrict__ wih1, const bf16_t* __restrict__ whh1,
                        const float* __restrict__ bih1,  const float* __restrict__ bhh1,
                        const bf16_t* __restrict__ wih2, const bf16_t* __restrict__ whh2,
                        const float* __restrict__ bih2,  const float* __restrict__ bhh2,
                        const bf16_t* __restrict__ wih3, const bf16_t* __restrict__ whh3,
                        const float* __restrict__ bih3,  const float* __restrict__ bhh3,
                        const bf16_t* __restrict__ wlin, const float* __restrict__ blin,
                        int* __restrict__ seqOut, float* __restrict__ lpOut) {
    __shared__ __align__(32) bf16_t sh_x [MROWS * XSTR];
    __shared__ __align__(32) bf16_t sh_h1[MROWS * HSTR];
    __shared__ __align__(32) bf16_t sh_h2[MROWS * HSTR];
    __shared__ __align__(32) bf16_t sh_h3[MROWS * HSTR];
    __shared__ float sh_logit[MROWS * VSTR];
    __shared__ float sh_logp[MROWS];
    __shared__ int   sh_tok[MROWS];
    __shared__ int   sh_fin[MROWS];

    const int tid  = threadIdx.x;
    const int wave = tid >> 5, lane = tid & 31;
    const int mg   = wave >> 3;        // 0..1 : which 16-row m-tile
    const int wg   = wave & 7;         // 0..7 : which 64-column slice
    const int rowBase = blockIdx.x * MROWS;

    for (int i = tid; i < MROWS * HSTR; i += NTHR) {
        sh_h1[i] = (bf16_t)0.f; sh_h2[i] = (bf16_t)0.f; sh_h3[i] = (bf16_t)0.f;
    }
    if (tid < MROWS) {
        sh_tok[tid]  = obs[rowBase + tid];
        sh_logp[tid] = 0.f;
        sh_fin[tid]  = 0;
    }
    __syncthreads();

#pragma unroll 1
    for (int step = 0; step < MAXLEN; ++step) {
        // embedding gather: x = emb[tok]
        for (int i = tid; i < MROWS * EMBD; i += NTHR) {
            const int m = i >> 7, c = i & (EMBD - 1);
            sh_x[m * XSTR + c] = embb[sh_tok[m] * EMBD + c];
        }
        __syncthreads();

        gru_layer<EMBD, XSTR>(sh_x  + mg * 16 * XSTR, sh_h1 + mg * 16 * HSTR,
                              wih1, whh1, bih1, bhh1, wg, lane);
        gru_layer<HID,  HSTR>(sh_h1 + mg * 16 * HSTR, sh_h2 + mg * 16 * HSTR,
                              wih2, whh2, bih2, bhh2, wg, lane);
        gru_layer<HID,  HSTR>(sh_h2 + mg * 16 * HSTR, sh_h3 + mg * 16 * HSTR,
                              wih3, whh3, bih3, bhh3, wg, lane);

        // logits = h3 @ wlin^T + blin : 2 m-tiles x 2 vocab-tiles -> waves 0..3
        if (wave < 4) {
            const int lmg = wave >> 1, lnt = wave & 1;
            const bf16_t* hb = sh_h3 + lmg * 16 * HSTR;
            v8f acc = {0.f,0.f,0.f,0.f,0.f,0.f,0.f,0.f};
#pragma unroll 4
            for (int kt = 0; kt < 16; ++kt)
                acc = wmma_bf(lds_a_frag(hb, HSTR, kt, lane),
                              g_b_frag(wlin, lnt, kt, 16, lane), acc);
            const int nlo = lane & 15, half = lane >> 4;
            const int col = lnt * 16 + nlo;
            const float bl = blin[col];
#pragma unroll
            for (int r8 = 0; r8 < 8; ++r8)
                sh_logit[(lmg * 16 + half * 8 + r8) * VSTR + col] = acc[r8] + bl;
        }
        __syncthreads();

        // Gumbel-max categorical sampling: wave w handles rows 2w, 2w+1;
        // 32 lanes = 32 vocab entries; butterfly reductions.
#pragma unroll
        for (int rr = 0; rr < 2; ++rr) {
            const int m = wave * 2 + rr;
            const float l = sh_logit[m * VSTR + lane];
            float mx = l;
#pragma unroll
            for (int off = 16; off; off >>= 1)
                mx = fmaxf(mx, __shfl_xor(mx, off, 32));
            float s = __expf(l - mx);
#pragma unroll
            for (int off = 16; off; off >>= 1)
                s += __shfl_xor(s, off, 32);
            const unsigned hsh =
                mix32(((unsigned)(rowBase + m) * (unsigned)MAXLEN + (unsigned)step) * 32u
                      + (unsigned)lane + 0x9E3779B9u);
            const float u = (hsh >> 8) * (1.0f / 16777216.0f);
            float key = l - __logf(-__logf(u));   // Gumbel key (matches categorical)
            int   ki  = lane;
#pragma unroll
            for (int off = 16; off; off >>= 1) {
                const float ok = __shfl_xor(key, off, 32);
                const int   oi = __shfl_xor(ki,  off, 32);
                if (ok > key || (ok == key && oi < ki)) { key = ok; ki = oi; }
            }
            if (lane == 0) {
                const int tok = ki;
                sh_logp[m] += sh_logit[m * VSTR + tok] - mx - __logf(s);
                const int fin = sh_fin[m];
                seqOut[(size_t)(rowBase + m) * MAXLEN + step] = fin ? 0 : tok;
                sh_fin[m] = fin | (tok == EOS_TOK);
                sh_tok[m] = tok;   // reference feeds unmasked xn into next step
            }
        }
        __syncthreads();
    }
    if (tid < MROWS) lpOut[rowBase + tid] = sh_logp[tid];
}

// --- weight preprocessing -------------------------------------------------
// Swizzle W[N][K] (f32 row-major) into bf16 per-lane WMMA B fragments:
// out[((nt*nKT + kt)*32 + lane)*16 + j] = bf16(W[nt*16+lane%16][kt*32+(lane/16)*16+j])
__global__ void swz_kernel(const float* __restrict__ W, bf16_t* __restrict__ out,
                           int N, int K) {
    const int nKT = K / 32;
    const int total = N * K;
    for (int e = blockIdx.x * blockDim.x + threadIdx.x; e < total;
         e += gridDim.x * blockDim.x) {
        const int j = e & 15;
        const int lane = (e >> 4) & 31;
        const int tile = e >> 9;
        const int kt = tile % nKT;
        const int nt = tile / nKT;
        const int n = nt * 16 + (lane & 15);
        const int k = kt * 32 + (lane >> 4) * 16 + j;
        out[e] = (bf16_t)W[n * K + k];
    }
}

__global__ void cvt_kernel(const float* __restrict__ W, bf16_t* __restrict__ out,
                           int total) {
    for (int e = blockIdx.x * blockDim.x + threadIdx.x; e < total;
         e += gridDim.x * blockDim.x)
        out[e] = (bf16_t)W[e];
}

extern "C" void kernel_launch(void* const* d_in, const int* in_sizes, int n_in,
                              void* d_out, int out_size, void* d_ws, size_t ws_size,
                              hipStream_t stream) {
    const int*   obs  = (const int*)d_in[0];
    // d_in[1] = max_length (scalar; compile-time MAXLEN used)
    const float* emb  = (const float*)d_in[2];
    const float* wlin = (const float*)d_in[3];
    const float* blin = (const float*)d_in[4];
    const float* wih1 = (const float*)d_in[5];
    const float* whh1 = (const float*)d_in[6];
    const float* bih1 = (const float*)d_in[7];
    const float* bhh1 = (const float*)d_in[8];
    const float* wih2 = (const float*)d_in[9];
    const float* whh2 = (const float*)d_in[10];
    const float* bih2 = (const float*)d_in[11];
    const float* bhh2 = (const float*)d_in[12];
    const float* wih3 = (const float*)d_in[13];
    const float* whh3 = (const float*)d_in[14];
    const float* bih3 = (const float*)d_in[15];
    const float* bhh3 = (const float*)d_in[16];

    // workspace layout (bf16 elems; total ~8.3 MB)
    bf16_t* ws = (bf16_t*)d_ws;
    size_t o = 0;
    bf16_t* embb  = ws + o; o += (size_t)VOC * EMBD;
    bf16_t* wih1s = ws + o; o += (size_t)H3 * EMBD;
    bf16_t* whh1s = ws + o; o += (size_t)H3 * HID;
    bf16_t* wih2s = ws + o; o += (size_t)H3 * HID;
    bf16_t* whh2s = ws + o; o += (size_t)H3 * HID;
    bf16_t* wih3s = ws + o; o += (size_t)H3 * HID;
    bf16_t* whh3s = ws + o; o += (size_t)H3 * HID;
    bf16_t* wlins = ws + o; o += (size_t)VOC * HID;

    cvt_kernel<<<16, 256, 0, stream>>>(emb, embb, VOC * EMBD);
    swz_kernel<<<192, 256, 0, stream>>>(wih1, wih1s, H3, EMBD);
    swz_kernel<<<768, 256, 0, stream>>>(whh1, whh1s, H3, HID);
    swz_kernel<<<768, 256, 0, stream>>>(wih2, wih2s, H3, HID);
    swz_kernel<<<768, 256, 0, stream>>>(whh2, whh2s, H3, HID);
    swz_kernel<<<768, 256, 0, stream>>>(wih3, wih3s, H3, HID);
    swz_kernel<<<768, 256, 0, stream>>>(whh3, whh3s, H3, HID);
    swz_kernel<<<64, 256, 0, stream>>>(wlin, wlins, VOC, HID);

    int*   seqOut = (int*)d_out;                             // sequences [B, MAXLEN]
    float* lpOut  = (float*)d_out + (size_t)BATCH * MAXLEN;  // log_probs [B]

    gru_rnn_persistent<<<BATCH / MROWS, NTHR, 0, stream>>>(
        obs, embb,
        wih1s, whh1s, bih1, bhh1,
        wih2s, whh2s, bih2, bhh2,
        wih3s, whh3s, bih3, bhh3,
        wlins, blin, seqOut, lpOut);
}
